// PyTorchMetalmLSTMBlock_10900626997372
// MI455X (gfx1250) — compile-verified, matrix operand order checked by reading, not run on hardware
//
#include <hip/hip_runtime.h>
#include <hip/hip_bf16.h>
#include <math.h>

typedef __attribute__((ext_vector_type(16))) _Float16 v16h;
typedef __attribute__((ext_vector_type(8)))  _Float16 v8h;
typedef __attribute__((ext_vector_type(8)))  float    v8f;

#define BB 4
#define SS 2048
#define DD 1024
#define HH 16
#define HD 64
#define BS (BB*SS)          // 8192 rows
#define CAPV 15.0f
#define EPSV 1e-6f

// ---------------------------------------------------------------------------
// WMMA fragment helpers (16x16x32 f16, wave32). Per ISA 7.12.2:
//   A (16x32): lane l<16 holds row M=l, K = hi*8+0..7 (s0..s7) and 16+hi*8+0..7
//   (s8..s15) where hi = lane>>4. B (32x16) loaded identically from W^T
//   (N-major), i.e. lane l holds column N=l with the same K enumeration.
//   C/D (16x16 f32): lane l holds N=l, VGPR r holds M = hi*8 + r.
// ---------------------------------------------------------------------------
__device__ inline v16h load_frag16(const _Float16* base, int ld, int row_base,
                                   int k_base, int lane) {
    int l = lane & 15, hi = lane >> 4;
    const _Float16* p = base + (size_t)(row_base + l) * ld + k_base + hi * 8;
    v8h g0 = *(const v8h*)(p);
    v8h g1 = *(const v8h*)(p + 16);
    v16h r;
#pragma unroll
    for (int i = 0; i < 8; ++i) { r[i] = g0[i]; r[i + 8] = g1[i]; }
    return r;
}

// ---------------------------------------------------------------------------
// 1) RMSNorm: xn_f16 = (ln_w * x * rsqrt(mean(x^2)+eps)) as f16
// ---------------------------------------------------------------------------
__global__ __launch_bounds__(256) void rmsnorm_k(const float* __restrict__ x,
                                                 const float* __restrict__ w,
                                                 _Float16* __restrict__ xn) {
    const int row = blockIdx.x;
    const int tid = threadIdx.x;
    const float* xr = x + (size_t)row * DD;
    float vals[4];
    float s = 0.f;
#pragma unroll
    for (int i = 0; i < 4; ++i) {
        vals[i] = xr[tid + i * 256];
        s += vals[i] * vals[i];
    }
#pragma unroll
    for (int m = 16; m >= 1; m >>= 1) s += __shfl_xor(s, m, 32);
    __shared__ float red[8];
    if ((tid & 31) == 0) red[tid >> 5] = s;
    __syncthreads();
    if (tid == 0) {
        float t = 0.f;
#pragma unroll
        for (int i = 0; i < 8; ++i) t += red[i];
        red[0] = rsqrtf(t / (float)DD + EPSV);
    }
    __syncthreads();
    const float rs = red[0];
    _Float16* xo = xn + (size_t)row * DD;
#pragma unroll
    for (int i = 0; i < 4; ++i) {
        int c = tid + i * 256;
        xo[c] = (_Float16)(vals[i] * rs * w[c]);
    }
}

// ---------------------------------------------------------------------------
// 2) Transpose+convert the four 1024x1024 weights to N-major f16 (LDS tiled)
// ---------------------------------------------------------------------------
__global__ __launch_bounds__(256) void transpose_w(
    const float* __restrict__ wq, const float* __restrict__ wk,
    const float* __restrict__ wv, const float* __restrict__ wo,
    _Float16* __restrict__ tq, _Float16* __restrict__ tk,
    _Float16* __restrict__ tv, _Float16* __restrict__ to) {
    const int z = blockIdx.z;
    const float* src = (z == 0) ? wq : (z == 1) ? wk : (z == 2) ? wv : wo;
    _Float16*   dst = (z == 0) ? tq : (z == 1) ? tk : (z == 2) ? tv : to;
    __shared__ float tile[32][33];
    const int bx = blockIdx.x * 32, by = blockIdx.y * 32;
    const int tx = threadIdx.x & 31, ty = threadIdx.x >> 5;  // ty 0..7
#pragma unroll
    for (int j = 0; j < 32; j += 8)
        tile[ty + j][tx] = src[(size_t)(by + ty + j) * DD + bx + tx];
    __syncthreads();
#pragma unroll
    for (int j = 0; j < 32; j += 8)
        dst[(size_t)(bx + ty + j) * DD + by + tx] = (_Float16)tile[tx][ty + j];
}

// ---------------------------------------------------------------------------
// 3) QKV GEMM: [8192,1024]f16 x W^T[1024,1024]f16 -> f16, f32 accumulate.
//    Block = 128 thr (4 waves); wave computes a 16x64 stripe; K-step 32.
// ---------------------------------------------------------------------------
__global__ __launch_bounds__(128) void gemm_qkv(
    const _Float16* __restrict__ xn,
    const _Float16* __restrict__ wtq, const _Float16* __restrict__ wtk,
    const _Float16* __restrict__ wtv,
    _Float16* __restrict__ q, _Float16* __restrict__ k,
    _Float16* __restrict__ v) {
    const int z = blockIdx.z;
    const _Float16* wt = (z == 0) ? wtq : (z == 1) ? wtk : wtv;
    _Float16* out      = (z == 0) ? q   : (z == 1) ? k   : v;
    const int n_tile = blockIdx.x * 64;
    const int m_base = blockIdx.y * 64 + (threadIdx.x >> 5) * 16;
    const int lane   = threadIdx.x & 31;
    v8f c0 = {}, c1 = {}, c2 = {}, c3 = {};
    for (int kk = 0; kk < DD; kk += 32) {
        if (kk + 32 < DD)
            __builtin_prefetch(xn + (size_t)(m_base + (lane & 15)) * DD + kk + 32, 0, 3);
        v16h a  = load_frag16(xn, DD, m_base, kk, lane);
        v16h b0 = load_frag16(wt, DD, n_tile +  0, kk, lane);
        v16h b1 = load_frag16(wt, DD, n_tile + 16, kk, lane);
        v16h b2 = load_frag16(wt, DD, n_tile + 32, kk, lane);
        v16h b3 = load_frag16(wt, DD, n_tile + 48, kk, lane);
        c0 = __builtin_amdgcn_wmma_f32_16x16x32_f16(false, a, false, b0, (short)0, c0, false, false);
        c1 = __builtin_amdgcn_wmma_f32_16x16x32_f16(false, a, false, b1, (short)0, c1, false, false);
        c2 = __builtin_amdgcn_wmma_f32_16x16x32_f16(false, a, false, b2, (short)0, c2, false, false);
        c3 = __builtin_amdgcn_wmma_f32_16x16x32_f16(false, a, false, b3, (short)0, c3, false, false);
    }
    const int l = lane & 15, hi = lane >> 4;
#pragma unroll
    for (int r = 0; r < 8; ++r) {
        const size_t m = (size_t)(m_base + hi * 8 + r);
        out[m * DD + n_tile +  0 + l] = (_Float16)c0[r];
        out[m * DD + n_tile + 16 + l] = (_Float16)c1[r];
        out[m * DD + n_tile + 32 + l] = (_Float16)c2[r];
        out[m * DD + n_tile + 48 + l] = (_Float16)c3[r];
    }
}

// ---------------------------------------------------------------------------
// 4) Gates: one wave per row; 16 outputs x 3 matrices; fp32 math,
//    softcap = 15*tanh(x/15), then sigmoid.
// ---------------------------------------------------------------------------
__global__ __launch_bounds__(32) void gates_k(
    const _Float16* __restrict__ xn, const float* __restrict__ Wi,
    const float* __restrict__ Wf, const float* __restrict__ Wo,
    float* __restrict__ ig, float* __restrict__ fg, float* __restrict__ og) {
    const int row = blockIdx.x;
    const int lane = threadIdx.x;
    float xv[32];
#pragma unroll
    for (int j = 0; j < 32; ++j)
        xv[j] = (float)xn[(size_t)row * DD + lane + j * 32];
    for (int g = 0; g < HH; ++g) {
        float ai = 0.f, af = 0.f, ao = 0.f;
#pragma unroll
        for (int j = 0; j < 32; ++j) {
            const int kk = lane + j * 32;
            const float xx = xv[j];
            ai += xx * Wi[kk * HH + g];
            af += xx * Wf[kk * HH + g];
            ao += xx * Wo[kk * HH + g];
        }
#pragma unroll
        for (int m = 16; m >= 1; m >>= 1) {
            ai += __shfl_xor(ai, m, 32);
            af += __shfl_xor(af, m, 32);
            ao += __shfl_xor(ao, m, 32);
        }
        if (lane == 0) {
            const size_t o = (size_t)row * HH + g;
            float ci = CAPV * tanhf(ai / CAPV);
            float cf = CAPV * tanhf(af / CAPV);
            float co = CAPV * tanhf(ao / CAPV);
            ig[o] = 1.f / (1.f + expf(-ci));
            fg[o] = 1.f / (1.f + expf(-cf));
            og[o] = 1.f / (1.f + expf(-co));
        }
    }
}

// ---------------------------------------------------------------------------
// 5) Sequential scan: one block per (b,h); 64x64 state in registers
//    (16 floats/lane, 256 lanes). Software-pipelined: while step t computes
//    out of LDS, step t+1's q/k/v/gates are in flight into registers.
//    kb/qb consumed as broadcast float4 (ds_load_b128) to cut DS ops 4x.
// ---------------------------------------------------------------------------
__global__ __launch_bounds__(256) void scan_k(
    const _Float16* __restrict__ q16, const _Float16* __restrict__ k16,
    const _Float16* __restrict__ v16f, const float* __restrict__ ig,
    const float* __restrict__ fg, const float* __restrict__ og,
    const float* __restrict__ hidden0, _Float16* __restrict__ h16,
    float* __restrict__ state_out) {
    const int b = blockIdx.x >> 4;
    const int h = blockIdx.x & 15;
    const int tid = threadIdx.x;
    const int e = tid & 63;
    const int d_base = (tid >> 6) * 16;
    const size_t sbase = ((size_t)b * HH + h) * HD * HD;
    const size_t col = (size_t)h * HD;
    float st[16];
#pragma unroll
    for (int r = 0; r < 16; ++r)
        st[r] = hidden0[sbase + (size_t)(d_base + r) * HD + e];

    __shared__ __align__(16) float kb[64];
    __shared__ __align__(16) float vb[64];
    __shared__ __align__(16) float qb[64];
    __shared__ float gg[3];
    __shared__ float part[256];

    // prologue: fetch t=0 into registers
    _Float16 kn = (_Float16)0.f, vn = (_Float16)0.f, qn = (_Float16)0.f;
    float gn = 0.f;
    {
        const size_t row0 = (size_t)b * SS;
        if (tid < 64) {
            const size_t idx = row0 * DD + col + tid;
            kn = k16[idx]; vn = v16f[idx]; qn = q16[idx];
        } else if (tid < 67) {
            const float* gp = (tid == 64) ? ig : (tid == 65) ? fg : og;
            gn = gp[row0 * HH + h];
        }
    }

    for (int t = 0; t < SS; ++t) {
        // publish step t operands
        if (tid < 64) {
            kb[tid] = (float)kn; vb[tid] = (float)vn; qb[tid] = (float)qn;
        } else if (tid < 67) {
            gg[tid - 64] = gn;
        }
        __syncthreads();
        // prefetch step t+1 into registers (consumed next iteration)
        if (t + 1 < SS) {
            const size_t row1 = (size_t)b * SS + t + 1;
            if (tid < 64) {
                const size_t idx = row1 * DD + col + tid;
                kn = k16[idx]; vn = v16f[idx]; qn = q16[idx];
            } else if (tid < 67) {
                const float* gp = (tid == 64) ? ig : (tid == 65) ? fg : og;
                gn = gp[row1 * HH + h];
            }
        }
        const float iG = gg[0], fG = gg[1], oG = gg[2];
        const float ve = vb[e];
        float acc = 0.f;
        const float4* kb4 = (const float4*)(kb + d_base);
        const float4* qb4 = (const float4*)(qb + d_base);
#pragma unroll
        for (int i = 0; i < 4; ++i) {
            const float4 kx = kb4[i];
            const float4 qx = qb4[i];
            float s0 = fG * st[4 * i + 0] + (iG * kx.x) * ve; st[4 * i + 0] = s0; acc += qx.x * s0;
            float s1 = fG * st[4 * i + 1] + (iG * kx.y) * ve; st[4 * i + 1] = s1; acc += qx.y * s1;
            float s2 = fG * st[4 * i + 2] + (iG * kx.z) * ve; st[4 * i + 2] = s2; acc += qx.z * s2;
            float s3 = fG * st[4 * i + 3] + (iG * kx.w) * ve; st[4 * i + 3] = s3; acc += qx.w * s3;
        }
        part[tid] = acc;
        __syncthreads();
        if (tid < 64) {
            const float hv =
                oG * (part[tid] + part[tid + 64] + part[tid + 128] + part[tid + 192]);
            h16[((size_t)b * SS + t) * DD + col + tid] = (_Float16)hv;
        }
        // next iteration's shared writes are ordered by its first barrier
    }
#pragma unroll
    for (int r = 0; r < 16; ++r)
        state_out[sbase + (size_t)(d_base + r) * HD + e] = st[r];
}

// ---------------------------------------------------------------------------
// 6) Output GEMM + residual: out = x + h @ Wout  (WMMA, fp32 store)
// ---------------------------------------------------------------------------
__global__ __launch_bounds__(128) void gemm_out_k(
    const _Float16* __restrict__ h16, const _Float16* __restrict__ woT,
    const float* __restrict__ x, float* __restrict__ out) {
    const int n_tile = blockIdx.x * 64;
    const int m_base = blockIdx.y * 64 + (threadIdx.x >> 5) * 16;
    const int lane   = threadIdx.x & 31;
    v8f c0 = {}, c1 = {}, c2 = {}, c3 = {};
    for (int kk = 0; kk < DD; kk += 32) {
        if (kk + 32 < DD)
            __builtin_prefetch(h16 + (size_t)(m_base + (lane & 15)) * DD + kk + 32, 0, 3);
        v16h a  = load_frag16(h16, DD, m_base, kk, lane);
        v16h b0 = load_frag16(woT, DD, n_tile +  0, kk, lane);
        v16h b1 = load_frag16(woT, DD, n_tile + 16, kk, lane);
        v16h b2 = load_frag16(woT, DD, n_tile + 32, kk, lane);
        v16h b3 = load_frag16(woT, DD, n_tile + 48, kk, lane);
        c0 = __builtin_amdgcn_wmma_f32_16x16x32_f16(false, a, false, b0, (short)0, c0, false, false);
        c1 = __builtin_amdgcn_wmma_f32_16x16x32_f16(false, a, false, b1, (short)0, c1, false, false);
        c2 = __builtin_amdgcn_wmma_f32_16x16x32_f16(false, a, false, b2, (short)0, c2, false, false);
        c3 = __builtin_amdgcn_wmma_f32_16x16x32_f16(false, a, false, b3, (short)0, c3, false, false);
    }
    const int l = lane & 15, hi = lane >> 4;
#pragma unroll
    for (int r = 0; r < 8; ++r) {
        const size_t m = (size_t)(m_base + hi * 8 + r);
        size_t i0 = m * DD + n_tile + l;
        out[i0 +  0] = x[i0 +  0] + c0[r];
        out[i0 + 16] = x[i0 + 16] + c1[r];
        out[i0 + 32] = x[i0 + 32] + c2[r];
        out[i0 + 48] = x[i0 + 48] + c3[r];
    }
}

// ---------------------------------------------------------------------------
extern "C" void kernel_launch(void* const* d_in, const int* in_sizes, int n_in,
                              void* d_out, int out_size, void* d_ws,
                              size_t ws_size, hipStream_t stream) {
    (void)in_sizes; (void)n_in; (void)out_size; (void)ws_size;
    const float* x       = (const float*)d_in[0];
    const float* hidden0 = (const float*)d_in[1];
    const float* Wq      = (const float*)d_in[2];
    const float* Wk      = (const float*)d_in[3];
    const float* Wv      = (const float*)d_in[4];
    const float* Wi      = (const float*)d_in[5];
    const float* Wf      = (const float*)d_in[6];
    const float* Wo      = (const float*)d_in[7];
    const float* Wout    = (const float*)d_in[8];
    const float* lnw     = (const float*)d_in[9];

    float* out       = (float*)d_out;
    float* state_out = out + (size_t)BS * DD;  // second tuple element

    char* ws = (char*)d_ws;
    _Float16* xn  = (_Float16*)ws; ws += (size_t)BS * DD * 2;
    _Float16* WqT = (_Float16*)ws; ws += (size_t)DD * DD * 2;
    _Float16* WkT = (_Float16*)ws; ws += (size_t)DD * DD * 2;
    _Float16* WvT = (_Float16*)ws; ws += (size_t)DD * DD * 2;
    _Float16* WoT = (_Float16*)ws; ws += (size_t)DD * DD * 2;
    _Float16* q16 = (_Float16*)ws; ws += (size_t)BS * DD * 2;
    _Float16* k16 = (_Float16*)ws; ws += (size_t)BS * DD * 2;
    _Float16* v16b= (_Float16*)ws; ws += (size_t)BS * DD * 2;
    _Float16* h16 = (_Float16*)ws; ws += (size_t)BS * DD * 2;
    float* ig = (float*)ws; ws += (size_t)BS * HH * 4;
    float* fg = (float*)ws; ws += (size_t)BS * HH * 4;
    float* og = (float*)ws; ws += (size_t)BS * HH * 4;

    rmsnorm_k<<<BS, 256, 0, stream>>>(x, lnw, xn);
    transpose_w<<<dim3(32, 32, 4), 256, 0, stream>>>(Wq, Wk, Wv, Wout,
                                                     WqT, WkT, WvT, WoT);
    gemm_qkv<<<dim3(DD / 64, BS / 64, 3), 128, 0, stream>>>(
        xn, WqT, WkT, WvT, q16, k16, v16b);
    gates_k<<<BS, 32, 0, stream>>>(xn, Wi, Wf, Wo, ig, fg, og);
    scan_k<<<BB * HH, 256, 0, stream>>>(q16, k16, v16b, ig, fg, og, hidden0,
                                        h16, state_out);
    gemm_out_k<<<dim3(DD / 64, BS / 64), 128, 0, stream>>>(h16, WoT, x, out);
}
